// LSTMDecoder_43602507989634
// MI455X (gfx1250) — compile-verified, matrix operand order checked by reading, not run on hardware
//
#include <hip/hip_runtime.h>
#include <math.h>

// ---------------------------------------------------------------------------
// Greedy LSTM decoder for MI455X (gfx1250, wave32, WMMA bf16 path).
// B=64, T=50, E=512, H=1024, V=32000.
// Weights converted once to bf16 (~78MB total -> resident in the 192MB L2).
// GEMMs use v_wmma_f32_16x16x32_bf16 with f32 accumulation.
// Tiling: each wave owns ONE weight N-tile and ALL FOUR M-tiles (M=64), so
// each weight fragment is loaded once per k-iter and reused by 4 WMMAs
// (weights are the dominant traffic; activations are WGP$-resident).
// ---------------------------------------------------------------------------

typedef __attribute__((ext_vector_type(16))) __bf16 v16bf;
typedef __attribute__((ext_vector_type(8)))  __bf16 v8bf;
typedef __attribute__((ext_vector_type(8)))  float  v8f;

static constexpr int Bq   = 64;     // batch
static constexpr int Tq   = 50;     // timesteps
static constexpr int Eq   = 512;    // embedding dim
static constexpr int Hq   = 1024;   // hidden dim
static constexpr int Vq   = 32000;  // vocab
static constexpr int Gq   = 4 * Hq; // 4096 gates

// ---- fp32 -> bf16 (round to nearest even) as raw ushort ----
static __device__ __forceinline__ unsigned short f2bf(float f) {
  union { float f; unsigned int u; } c; c.f = f;
  unsigned int u = c.u;
  unsigned int r = u + 0x7FFFu + ((u >> 16) & 1u);
  if ((u & 0x7FFFFFFFu) > 0x7F800000u) r = u | 0x00400000u; // quiet NaN
  return (unsigned short)(r >> 16);
}

// ---------------------------------------------------------------------------
// Load a 16x32 bf16 WMMA fragment (A layout; B uses the same mapping with the
// N index in place of M since W is row-major [N,K] and B(k,n)=W[n][k]).
// Lane L: row = row0 + (L%16); K-run = k0 + (L/16)*8 + {0..7} and +16.
// Two aligned 16-byte loads per lane -> global_load_b128 pairs.
// ---------------------------------------------------------------------------
static __device__ __forceinline__ v16bf load_frag(const unsigned short* base,
                                                  int row0, int ld, int k0) {
  int lane = threadIdx.x & 31;
  const unsigned short* p =
      base + (size_t)(row0 + (lane & 15)) * ld + k0 + ((lane >> 4) << 3);
  v8bf lo = *(const v8bf*)p;
  v8bf hi = *(const v8bf*)(p + 16);
  return __builtin_shufflevector(lo, hi, 0, 1, 2, 3, 4, 5, 6, 7,
                                 8, 9, 10, 11, 12, 13, 14, 15);
}

// ---------------------------------------------------------------------------
// Prepass: fp32 weights -> bf16 workspace
// ---------------------------------------------------------------------------
__global__ __launch_bounds__(256) void f32_to_bf16_kernel(const float* __restrict__ src,
                                                          unsigned short* __restrict__ dst,
                                                          int n) {
  int i = blockIdx.x * 256 + threadIdx.x;
  if (i < n) dst[i] = f2bf(src[i]);
}

// Init: c = 0, h = bf16(encoder embeddings), x = bf16(emb_table[SOS=0]) per row
__global__ __launch_bounds__(256) void init_state_kernel(const float* __restrict__ embeddings,
                                                         const float* __restrict__ emb_table,
                                                         float* __restrict__ c,
                                                         unsigned short* __restrict__ hbf,
                                                         unsigned short* __restrict__ xbf) {
  int i = blockIdx.x * 256 + threadIdx.x;
  if (i < Bq * Hq) {
    c[i] = 0.0f;
    hbf[i] = f2bf(embeddings[i]);
  }
  if (i < Bq * Eq) {
    xbf[i] = f2bf(emb_table[i % Eq]); // SOS row 0 broadcast over batch
  }
}

// ---------------------------------------------------------------------------
// gates[64,4096] = x_bf @ W_ih^T + h_bf @ W_hh^T + b_ih + b_hh
// block = 128 (4 waves); each wave: one N tile (16 gates), all 4 M tiles.
// Weight fragment loaded once per k-iter, reused across 4 WMMAs.
// grid = 4096 / 16 / 4 = 64 blocks.
// ---------------------------------------------------------------------------
__global__ __launch_bounds__(128) void gates_wmma_kernel(
    const unsigned short* __restrict__ xbf, const unsigned short* __restrict__ hbf,
    const unsigned short* __restrict__ Wih, const unsigned short* __restrict__ Whh,
    const float* __restrict__ bih, const float* __restrict__ bhh,
    float* __restrict__ gates) {
  const int lane = threadIdx.x & 31;
  const int n0 = (blockIdx.x * 4 + (threadIdx.x >> 5)) * 16;

  v8f acc0 = {}, acc1 = {}, acc2 = {}, acc3 = {};
#pragma unroll 2
  for (int k = 0; k < Eq; k += 32) {
    v16bf b  = load_frag(Wih, n0, Eq, k);      // weights: load once
    v16bf a0 = load_frag(xbf,  0, Eq, k);      // activations: WGP$-resident
    v16bf a1 = load_frag(xbf, 16, Eq, k);
    v16bf a2 = load_frag(xbf, 32, Eq, k);
    v16bf a3 = load_frag(xbf, 48, Eq, k);
    acc0 = __builtin_amdgcn_wmma_f32_16x16x32_bf16(false, a0, false, b, (short)0, acc0, false, false);
    acc1 = __builtin_amdgcn_wmma_f32_16x16x32_bf16(false, a1, false, b, (short)0, acc1, false, false);
    acc2 = __builtin_amdgcn_wmma_f32_16x16x32_bf16(false, a2, false, b, (short)0, acc2, false, false);
    acc3 = __builtin_amdgcn_wmma_f32_16x16x32_bf16(false, a3, false, b, (short)0, acc3, false, false);
  }
#pragma unroll 2
  for (int k = 0; k < Hq; k += 32) {
    v16bf b  = load_frag(Whh, n0, Hq, k);
    v16bf a0 = load_frag(hbf,  0, Hq, k);
    v16bf a1 = load_frag(hbf, 16, Hq, k);
    v16bf a2 = load_frag(hbf, 32, Hq, k);
    v16bf a3 = load_frag(hbf, 48, Hq, k);
    acc0 = __builtin_amdgcn_wmma_f32_16x16x32_bf16(false, a0, false, b, (short)0, acc0, false, false);
    acc1 = __builtin_amdgcn_wmma_f32_16x16x32_bf16(false, a1, false, b, (short)0, acc1, false, false);
    acc2 = __builtin_amdgcn_wmma_f32_16x16x32_bf16(false, a2, false, b, (short)0, acc2, false, false);
    acc3 = __builtin_amdgcn_wmma_f32_16x16x32_bf16(false, a3, false, b, (short)0, acc3, false, false);
  }

  const int n = n0 + (lane & 15);
  const float bias = bih[n] + bhh[n];
  const int mbase = (lane >> 4) << 3; // C/D layout: M = 8*(lane/16)+r
#pragma unroll
  for (int r = 0; r < 8; ++r) {
    gates[(size_t)(mbase + r +  0) * Gq + n] = acc0[r] + bias;
    gates[(size_t)(mbase + r + 16) * Gq + n] = acc1[r] + bias;
    gates[(size_t)(mbase + r + 32) * Gq + n] = acc2[r] + bias;
    gates[(size_t)(mbase + r + 48) * Gq + n] = acc3[r] + bias;
  }
}

// ---------------------------------------------------------------------------
// LSTM cell: PyTorch gate order [i, f, g, o]; updates c (f32), emits h (bf16)
// ---------------------------------------------------------------------------
__global__ __launch_bounds__(256) void lstm_cell_kernel(const float* __restrict__ gates,
                                                        float* __restrict__ c,
                                                        unsigned short* __restrict__ hbf) {
  int idx = blockIdx.x * 256 + threadIdx.x;
  if (idx >= Bq * Hq) return;
  int b = idx >> 10;           // / Hq
  int j = idx & (Hq - 1);      // % Hq
  const float* g = gates + (size_t)b * Gq;
  float gi = g[j];
  float gf = g[j + Hq];
  float gg = g[j + 2 * Hq];
  float go = g[j + 3 * Hq];
  float si = 1.0f / (1.0f + __expf(-gi));
  float sf = 1.0f / (1.0f + __expf(-gf));
  float so = 1.0f / (1.0f + __expf(-go));
  float cn = sf * c[idx] + si * tanhf(gg);
  float hn = so * tanhf(cn);
  c[idx] = cn;
  hbf[idx] = f2bf(hn);
}

// ---------------------------------------------------------------------------
// logits[b, t, :] = h_bf @ W_out^T + b_out, written straight into d_out
// (layout [B, T, V]); each wave: one vocab N-tile, all 4 M tiles.
// grid = 32000 / 16 / 4 = 500 blocks. Minimal W_out L2 traffic (64MB/step).
// ---------------------------------------------------------------------------
__global__ __launch_bounds__(128) void logits_wmma_kernel(
    const unsigned short* __restrict__ hbf, const unsigned short* __restrict__ Wout,
    const float* __restrict__ bout, float* __restrict__ out_logits, int t) {
  const int lane = threadIdx.x & 31;
  const int n0 = (blockIdx.x * 4 + (threadIdx.x >> 5)) * 16;

  v8f acc0 = {}, acc1 = {}, acc2 = {}, acc3 = {};
#pragma unroll 2
  for (int k = 0; k < Hq; k += 32) {
    v16bf b  = load_frag(Wout, n0, Hq, k);
    v16bf a0 = load_frag(hbf,  0, Hq, k);
    v16bf a1 = load_frag(hbf, 16, Hq, k);
    v16bf a2 = load_frag(hbf, 32, Hq, k);
    v16bf a3 = load_frag(hbf, 48, Hq, k);
    acc0 = __builtin_amdgcn_wmma_f32_16x16x32_bf16(false, a0, false, b, (short)0, acc0, false, false);
    acc1 = __builtin_amdgcn_wmma_f32_16x16x32_bf16(false, a1, false, b, (short)0, acc1, false, false);
    acc2 = __builtin_amdgcn_wmma_f32_16x16x32_bf16(false, a2, false, b, (short)0, acc2, false, false);
    acc3 = __builtin_amdgcn_wmma_f32_16x16x32_bf16(false, a3, false, b, (short)0, acc3, false, false);
  }

  const int n = n0 + (lane & 15);
  const float bias = bout[n];
  const int mbase = (lane >> 4) << 3;
#pragma unroll
  for (int r = 0; r < 8; ++r) {
    out_logits[((size_t)(mbase + r +  0) * Tq + t) * Vq + n] = acc0[r] + bias;
    out_logits[((size_t)(mbase + r + 16) * Tq + t) * Vq + n] = acc1[r] + bias;
    out_logits[((size_t)(mbase + r + 32) * Tq + t) * Vq + n] = acc2[r] + bias;
    out_logits[((size_t)(mbase + r + 48) * Tq + t) * Vq + n] = acc3[r] + bias;
  }
}

// ---------------------------------------------------------------------------
// Per-batch-row argmax over V=32000 (first-occurrence tie-break), write pred
// as float into d_out preds region, gather next input embedding as bf16.
// ---------------------------------------------------------------------------
__global__ __launch_bounds__(256) void argmax_gather_kernel(
    const float* __restrict__ out_logits, const float* __restrict__ emb_table,
    float* __restrict__ preds, unsigned short* __restrict__ xbf, int t) {
  __shared__ float sval[256];
  __shared__ int   sidx[256];
  __shared__ int   s_best;

  const int b = blockIdx.x;
  const float* row = out_logits + ((size_t)b * Tq + t) * Vq;

  float best = -INFINITY;
  int bidx = 0;
  for (int v = threadIdx.x; v < Vq; v += 256) {
    float x = row[v];
    if (x > best) { best = x; bidx = v; }
  }
  sval[threadIdx.x] = best;
  sidx[threadIdx.x] = bidx;
  __syncthreads();

  for (int s = 128; s > 0; s >>= 1) {
    if ((int)threadIdx.x < s) {
      float ov = sval[threadIdx.x + s];
      int   oi = sidx[threadIdx.x + s];
      float mv = sval[threadIdx.x];
      int   mi = sidx[threadIdx.x];
      if (ov > mv || (ov == mv && oi < mi)) {
        sval[threadIdx.x] = ov;
        sidx[threadIdx.x] = oi;
      }
    }
    __syncthreads();
  }

  if (threadIdx.x == 0) {
    s_best = sidx[0];
    preds[b * Tq + t] = (float)sidx[0];
  }
  __syncthreads();

  const float* erow = emb_table + (size_t)s_best * Eq;
  for (int e = threadIdx.x; e < Eq; e += 256) {
    xbf[(size_t)b * Eq + e] = f2bf(erow[e]);
  }
}

// ---------------------------------------------------------------------------
// Host-side orchestration (graph-capture safe: launches only).
// ---------------------------------------------------------------------------
extern "C" void kernel_launch(void* const* d_in, const int* in_sizes, int n_in,
                              void* d_out, int out_size, void* d_ws, size_t ws_size,
                              hipStream_t stream) {
  // inputs per setup_inputs() order
  const float* embeddings = (const float*)d_in[1];
  const float* emb_table  = (const float*)d_in[2];
  const float* W_ih       = (const float*)d_in[3];
  const float* W_hh       = (const float*)d_in[4];
  const float* b_ih       = (const float*)d_in[5];
  const float* b_hh       = (const float*)d_in[6];
  const float* W_out      = (const float*)d_in[7];
  const float* b_out      = (const float*)d_in[8];
  (void)in_sizes; (void)n_in; (void)out_size; (void)ws_size;

  // workspace layout (256B-aligned offsets)
  char* ws = (char*)d_ws;
  unsigned short* Wih_bf  = (unsigned short*)(ws + 0);
  unsigned short* Whh_bf  = (unsigned short*)(ws + 4194304);   // + 4096*512*2
  unsigned short* Wout_bf = (unsigned short*)(ws + 12582912);  // + 4096*1024*2
  float*          gates   = (float*)(ws + 78118912);           // + 32000*1024*2
  float*          c_state = (float*)(ws + 79167488);           // + 64*4096*4
  unsigned short* h_bf    = (unsigned short*)(ws + 79429632);  // + 64*1024*4
  unsigned short* x_bf    = (unsigned short*)(ws + 79560704);  // + 64*1024*2

  float* preds      = (float*)d_out;           // [B, T] as float
  float* out_logits = (float*)d_out + Bq * Tq; // [B, T, V] f32

  // --- prepass: weight conversion + state init ---
  {
    int n1 = Gq * Eq;                 // 2,097,152
    int n2 = Gq * Hq;                 // 4,194,304
    int n3 = Vq * Hq;                 // 32,768,000
    f32_to_bf16_kernel<<<(n1 + 255) / 256, 256, 0, stream>>>(W_ih, Wih_bf, n1);
    f32_to_bf16_kernel<<<(n2 + 255) / 256, 256, 0, stream>>>(W_hh, Whh_bf, n2);
    f32_to_bf16_kernel<<<(n3 + 255) / 256, 256, 0, stream>>>(W_out, Wout_bf, n3);
    init_state_kernel<<<(Bq * Hq + 255) / 256, 256, 0, stream>>>(
        embeddings, emb_table, c_state, h_bf, x_bf);
  }

  // --- 50 sequential decode steps ---
  for (int t = 0; t < Tq; ++t) {
    gates_wmma_kernel<<<Gq / 64, 128, 0, stream>>>(x_bf, h_bf, Wih_bf, Whh_bf,
                                                   b_ih, b_hh, gates);
    lstm_cell_kernel<<<(Bq * Hq + 255) / 256, 256, 0, stream>>>(gates, c_state, h_bf);
    logits_wmma_kernel<<<Vq / 64, 128, 0, stream>>>(h_bf, Wout_bf, b_out,
                                                    out_logits, t);
    argmax_gather_kernel<<<Bq, 256, 0, stream>>>(out_logits, emb_table,
                                                 preds, x_bf, t);
  }
}
